// TextureFeatureExtractor_73229192397344
// MI455X (gfx1250) — compile-verified, matrix operand order checked by reading, not run on hardware
//
#include <hip/hip_runtime.h>

typedef __attribute__((ext_vector_type(16))) _Float16 v16h;
typedef __attribute__((ext_vector_type(8)))  float    v8f;

#define BATCH 32
#define H     256
#define W     256
#define NPIX  (H*W)
#define NF    32          // number of gabor filters
#define KS    31          // gabor kernel size
#define KK    (KS*KS)     // 961 taps
#define NSTEP 31          // K-steps: one filter row per 16x16x32 WMMA step (kx 0..30 + 1 pad)
#define NBINS 26

#define STRIPW 159        // 128 pixels + 15 halo left + 16 halo right (incl. pad tap col)
#define STRIDE 160        // LDS row stride in halfs

// ---------------------------------------------------------------------------
// 1) RGB -> rounded grayscale (cv2 semantics), stored as fp16 (0..255 exact)
// ---------------------------------------------------------------------------
__global__ void prep_gray(const float* __restrict__ x, _Float16* __restrict__ gray16) {
    int i = blockIdx.x * blockDim.x + threadIdx.x;
    if (i >= BATCH * NPIX) return;
    int b = i / NPIX, p = i - b * NPIX;
    const float* base = x + (size_t)b * 3 * NPIX;
    float r  = rintf(base[p] * 255.0f);
    float g  = rintf(base[NPIX + p] * 255.0f);
    float bl = rintf(base[2 * NPIX + p] * 255.0f);
    float gv = rintf(0.299f * r + 0.587f * g + 0.114f * bl);
    gray16[i] = (_Float16)gv;
}

__global__ void zero_f32(float* __restrict__ p, int n) {
    int i = blockIdx.x * blockDim.x + threadIdx.x;
    if (i < n) p[i] = 0.0f;
}

// ---------------------------------------------------------------------------
// 2a) Pack the filter bank into WMMA B-fragment order, once per launch.
//     Layout: Bfrag[((s*2 + ntile)*32 + lane)*16 + e], fp16.
//     K-permutation: step s = filter row, kx = e + 8*(e>=8) + 8*(lane>>4),
//     kx==31 is a zero pad tap.  Each lane's fragment = 32 contiguous bytes.
// ---------------------------------------------------------------------------
__global__ void pack_bfrag(const float* __restrict__ gab,      // [32][961]
                           _Float16* __restrict__ Bfrag) {     // [31*2*32*16]
    int idx = blockIdx.x * blockDim.x + threadIdx.x;
    if (idx >= NSTEP * 2 * 32 * 16) return;
    int e    = idx & 15;
    int lane = (idx >> 4) & 31;
    int nt   = (idx >> 9) & 1;
    int s    = idx >> 10;
    int kx   = e + ((e >= 8) ? 8 : 0) + 8 * (lane >> 4);
    int n    = nt * 16 + (lane & 15);
    float v  = (kx < KS) ? gab[n * KK + s * KS + kx] : 0.0f;
    Bfrag[idx] = (_Float16)v;
}

// ---------------------------------------------------------------------------
// 2b) Gabor bank on the WMMA pipe.
//     Workgroup = 8 waves = 8 adjacent 16-pixel tiles on one image row.
//     Reflect-101 handled once while staging a 31x159 fp16 strip into LDS.
//     Inner loop: A = 2x ds_load_b128, B = 2x global_load_b128 per fragment,
//     4 independent WMMA accumulation chains (even/odd K-steps x 2 N-tiles)
//     to avoid accumulator-rotation copies and WMMA hazard NOPs.
// ---------------------------------------------------------------------------
__device__ __forceinline__ v16h load_afrag(const _Float16* __restrict__ ap) {
    v16h a;
#pragma unroll
    for (int e = 0; e < 8; ++e)  a[e] = ap[e];        // kx = 8*half + e
#pragma unroll
    for (int e = 8; e < 16; ++e) a[e] = ap[8 + e];    // kx = 16 + 8*half + (e-8)
    return a;
}

__launch_bounds__(256)
__global__ void gabor_wmma(const _Float16* __restrict__ gray16,
                           const _Float16* __restrict__ Bfrag,
                           float*          __restrict__ sums)  // [B*NF] sum, [B*NF] sumsq
{
    __shared__ _Float16 strip[NSTEP * STRIDE];   // ~9.9 KB

    // tile group: b, row y, xg in {0,128}
    const int b   = blockIdx.x >> 9;
    const int rem = blockIdx.x & 511;
    const int y   = rem >> 1;
    const int xg  = (rem & 1) << 7;

    const _Float16* gbase = gray16 + (size_t)b * NPIX;

    // stage reflect-101 padded strip: rows y-15..y+15, cols xg-15..xg+143
    for (int i = threadIdx.x; i < NSTEP * STRIPW; i += 256) {
        int s = i / STRIPW;
        int c = i - s * STRIPW;
        int sy = y - 15 + s;
        int sx = xg - 15 + c;
        sy = (sy < 0) ? -sy : ((sy > 255) ? 510 - sy : sy);
        sx = (sx < 0) ? -sx : ((sx > 255) ? 510 - sx : sx);
        strip[s * STRIDE + c] = gbase[sy * W + sx];
    }
    __syncthreads();

    const int wave = threadIdx.x >> 5;
    const int lane = threadIdx.x & 31;
    const int half = lane >> 4;                    // fragment K-half
    const int mn   = lane & 15;                    // M (pixel) for A, N (filter) for B/C
    const int abase0 = wave * 16 + mn + 8 * half;  // strip column base for this lane

    const v16h* bptr = (const v16h*)Bfrag;         // 32B-aligned fragments

    // 4 independent accumulation chains: (even, odd) x (filters 0-15, 16-31)
    v8f c0a = {}, c0b = {}, c1a = {}, c1b = {};
    for (int s = 0; s < NSTEP - 1; s += 2) {
        v16h aE  = load_afrag(strip + (s)     * STRIDE + abase0);
        v16h aO  = load_afrag(strip + (s + 1) * STRIDE + abase0);
        v16h b0E = bptr[((s)     * 2 + 0) * 32 + lane];
        v16h b1E = bptr[((s)     * 2 + 1) * 32 + lane];
        v16h b0O = bptr[((s + 1) * 2 + 0) * 32 + lane];
        v16h b1O = bptr[((s + 1) * 2 + 1) * 32 + lane];
        c0a = __builtin_amdgcn_wmma_f32_16x16x32_f16(false, aE, false, b0E, (short)0, c0a, false, false);
        c1a = __builtin_amdgcn_wmma_f32_16x16x32_f16(false, aE, false, b1E, (short)0, c1a, false, false);
        c0b = __builtin_amdgcn_wmma_f32_16x16x32_f16(false, aO, false, b0O, (short)0, c0b, false, false);
        c1b = __builtin_amdgcn_wmma_f32_16x16x32_f16(false, aO, false, b1O, (short)0, c1b, false, false);
    }
    {   // tail step s = 30
        v16h aT  = load_afrag(strip + (NSTEP - 1) * STRIDE + abase0);
        v16h b0T = bptr[((NSTEP - 1) * 2 + 0) * 32 + lane];
        v16h b1T = bptr[((NSTEP - 1) * 2 + 1) * 32 + lane];
        c0a = __builtin_amdgcn_wmma_f32_16x16x32_f16(false, aT, false, b0T, (short)0, c0a, false, false);
        c1a = __builtin_amdgcn_wmma_f32_16x16x32_f16(false, aT, false, b1T, (short)0, c1a, false, false);
    }

    // merge chains, then reduce over M (C layout: lane = column n, VGPR r = row)
    float s0 = 0.f, q0 = 0.f, s1 = 0.f, q1 = 0.f;
#pragma unroll
    for (int r = 0; r < 8; ++r) {
        float v0 = c0a[r] + c0b[r];
        float v1 = c1a[r] + c1b[r];
        s0 += v0; q0 += v0 * v0;
        s1 += v1; q1 += v1 * v1;
    }
    s0 += __shfl_xor(s0, 16); q0 += __shfl_xor(q0, 16);
    s1 += __shfl_xor(s1, 16); q1 += __shfl_xor(q1, 16);
    if (half == 0) {
        atomicAdd(&sums[b * NF + mn],                   s0);
        atomicAdd(&sums[BATCH * NF + b * NF + mn],      q0);
        atomicAdd(&sums[b * NF + 16 + mn],              s1);
        atomicAdd(&sums[BATCH * NF + b * NF + 16 + mn], q1);
    }
}

// ---------------------------------------------------------------------------
// 3) Uniform LBP (P=24, R=3), zero padding, LDS histogram per block.
// ---------------------------------------------------------------------------
__device__ const float LBP_DR[24] = {
     0.0f,    -0.77646f, -1.5f,     -2.12132f, -2.59808f, -2.89778f,
    -3.0f,    -2.89778f, -2.59808f, -2.12132f, -1.5f,     -0.77646f,
     0.0f,     0.77646f,  1.5f,      2.12132f,  2.59808f,  2.89778f,
     3.0f,     2.89778f,  2.59808f,  2.12132f,  1.5f,      0.77646f };
__device__ const float LBP_DC[24] = {
     3.0f,     2.89778f,  2.59808f,  2.12132f,  1.5f,      0.77646f,
     0.0f,    -0.77646f, -1.5f,     -2.12132f, -2.59808f, -2.89778f,
    -3.0f,    -2.89778f, -2.59808f, -2.12132f, -1.5f,     -0.77646f,
     0.0f,     0.77646f,  1.5f,      2.12132f,  2.59808f,  2.89778f };

__device__ __forceinline__ float ld_zero(const _Float16* g, int r, int c) {
    if (r < 0 || r >= H || c < 0 || c >= W) return 0.0f;
    return (float)g[r * W + c];
}

__launch_bounds__(256)
__global__ void lbp_kernel(const _Float16* __restrict__ gray16,
                           float* __restrict__ hist)           // [B][26]
{
    __shared__ int lh[NBINS];
    if (threadIdx.x < NBINS) lh[threadIdx.x] = 0;
    __syncthreads();

    int i = blockIdx.x * 256 + threadIdx.x;       // NPIX % 256 == 0 -> one image/block
    int b = i / NPIX, p = i - b * NPIX;
    int y = p >> 8, x = p & 255;
    const _Float16* g = gray16 + (size_t)b * NPIX;
    float center = (float)g[y * W + x];

    int bits[24];
#pragma unroll
    for (int pp = 0; pp < 24; ++pp) {
        float drv = LBP_DR[pp], dcv = LBP_DC[pp];
        float frf = floorf(drv), fcf = floorf(dcv);
        int   fr  = (int)frf,    fc  = (int)fcf;
        float tr  = drv - frf,   tc  = dcv - fcf;
        float v00 = ld_zero(g, y + fr,     x + fc);
        float v01 = ld_zero(g, y + fr,     x + fc + 1);
        float v10 = ld_zero(g, y + fr + 1, x + fc);
        float v11 = ld_zero(g, y + fr + 1, x + fc + 1);
        float samp = (1.f - tr) * (1.f - tc) * v00 + (1.f - tr) * tc * v01
                   + tr * (1.f - tc) * v10 + tr * tc * v11;
        bits[pp] = (samp >= center) ? 1 : 0;
    }
    int ones = 0, trans = 0;
#pragma unroll
    for (int pp = 0; pp < 24; ++pp) {
        ones  += bits[pp];
        trans += (bits[pp] != bits[(pp + 1) % 24]) ? 1 : 0;
    }
    int code = (trans <= 2) ? ones : 25;
    atomicAdd(&lh[code], 1);
    __syncthreads();
    if (threadIdx.x < NBINS) {
        int c = lh[threadIdx.x];
        if (c) atomicAdd(&hist[b * NBINS + threadIdx.x], (float)c);
    }
}

// ---------------------------------------------------------------------------
// 4) Finalize: interleaved (mean,std) x32 then normalized 26-bin histogram.
// ---------------------------------------------------------------------------
__global__ void finalize(const float* __restrict__ sums,
                         const float* __restrict__ hist,
                         float* __restrict__ out)              // [B][90]
{
    int t = threadIdx.x;                                       // 1024 threads
    const float inv = 1.0f / 65536.0f;
    if (t < BATCH * NF) {
        int b = t >> 5, f = t & 31;
        float mean = sums[t] * inv;
        float var  = sums[BATCH * NF + t] * inv - mean * mean;
        float sd   = sqrtf(fmaxf(var, 0.0f));
        out[b * 90 + 2 * f]     = mean;
        out[b * 90 + 2 * f + 1] = sd;
    }
    if (t < BATCH * NBINS) {
        int b = t / NBINS, j = t - b * NBINS;
        out[b * 90 + 64 + j] = hist[t] * inv;                  // sum is exactly 65536
    }
}

// ---------------------------------------------------------------------------
extern "C" void kernel_launch(void* const* d_in, const int* in_sizes, int n_in,
                              void* d_out, int out_size, void* d_ws, size_t ws_size,
                              hipStream_t stream) {
    (void)in_sizes; (void)n_in; (void)out_size; (void)ws_size;
    const float* x   = (const float*)d_in[0];   // [32,3,256,256]
    const float* gab = (const float*)d_in[1];   // [32,31,31]
    float* out = (float*)d_out;                 // [32,90]

    char* ws = (char*)d_ws;
    _Float16* gray16 = (_Float16*)ws;                                    // 4 MiB
    _Float16* Bfrag  = (_Float16*)(ws + (size_t)BATCH * NPIX * 2);       // 62 KB (32B aligned)
    float*    sums   = (float*)(ws + (size_t)BATCH * NPIX * 2
                                   + (size_t)NSTEP * 2 * 32 * 16 * 2);   // 2048 f32
    float*    hist   = sums + 2 * BATCH * NF;                            // 832 f32 (contiguous)

    prep_gray<<<(BATCH * NPIX) / 256, 256, 0, stream>>>(x, gray16);
    pack_bfrag<<<(NSTEP * 2 * 32 * 16 + 255) / 256, 256, 0, stream>>>(gab, Bfrag);
    zero_f32<<<(2 * BATCH * NF + BATCH * NBINS + 255) / 256, 256, 0, stream>>>(
        sums, 2 * BATCH * NF + BATCH * NBINS);
    gabor_wmma<<<BATCH * H * 2, 256, 0, stream>>>(gray16, Bfrag, sums);
    lbp_kernel<<<(BATCH * NPIX) / 256, 256, 0, stream>>>(gray16, hist);
    finalize<<<1, 1024, 0, stream>>>(sums, hist, out);
}
